// TemporalAttention_44238163149211
// MI455X (gfx1250) — matrix operand order repaired and checked by reading, not compiled, not run on hardware
//
#include <hip/hip_runtime.h>
#include <hip/hip_bf16.h>

// ---------------------------------------------------------------------------
// TemporalAttention fused kernel for MI455X (gfx1250, wave32, WMMA)
//
// B=8192, N=32 neighbors, node=172, edge=172, time=100
// KV GEMM: [B*N, 444] @ Wkv^T[444,256]  (dominant: ~60 GFLOP)
// Fused: KV-proj (WMMA f16) -> Z tile in LDS -> attention+softmax ->
//        O-proj -> LayerNorm, one workgroup per 4 batch rows.
// ---------------------------------------------------------------------------

#define BATCH      8192
#define NNBR       32
#define NODE_D     172
#define EDGE_D     172
#define TIME_D     100
#define KKV        444            // node+edge+time
#define KKV_PAD    448            // 14 chunks of 32
#define KQ         272            // node+time
#define OUTD       128
#define NHEADS     2
#define HEADD      64
#define LN_EPS     1e-5f

#define BPERWG     4              // batch rows per workgroup
#define ROWS       (BPERWG*NNBR)  // 128 KV rows per workgroup
#define ZS         260            // Z tile row stride (floats), padded
#define ASTR       34             // A staging row stride (halfs), padded
#define BSTR       34             // B staging row stride (halfs), padded

#define Z_BYTES    (ROWS*ZS*4)              // 133120
#define AS_BYTES   (ROWS*ASTR*2)            // 8704
#define BS_BYTES   (256*BSTR*2)             // 17408
#define STAGE_BYTES (AS_BYTES+BS_BYTES)     // 26112
#define SMEM_BYTES (Z_BYTES+STAGE_BYTES)    // 159232

typedef __attribute__((ext_vector_type(16))) _Float16 v16h;
typedef __attribute__((ext_vector_type(8)))  float    v8f;

// ---- weight prep: Wkv[256][444] f32 -> ws [256][448] f16 (zero padded) ----
__global__ void prep_wkv_f16(const float* __restrict__ Wkv,
                             _Float16* __restrict__ WkvH) {
  int idx = blockIdx.x * 256 + threadIdx.x;          // 256*448 total
  int n = idx / KKV_PAD, k = idx - n * KKV_PAD;
  float v = (k < KKV) ? Wkv[n * KKV + k] : 0.0f;
  WkvH[idx] = (_Float16)v;
}

// ---- fragment builders (contiguous k = 16*half + s for A and B, wave32) ----
__device__ __forceinline__ v16h load_a_frag(const _Float16* As, int row, int half) {
  union { unsigned u[8]; v16h v; } x;
  const unsigned* p = (const unsigned*)(As + row * ASTR);
#pragma unroll
  for (int i = 0; i < 8; ++i) {
    int koff = 2 * i + half * 16; // even
    x.u[i] = p[koff >> 1];
  }
  return x.v;
}

__device__ __forceinline__ v16h load_b_frag(const _Float16* Bs, int col, int kq) {
  union { unsigned u[8]; v16h v; } x;
  const unsigned* p = (const unsigned*)(Bs + col * BSTR + kq); // kq in {0,16}
#pragma unroll
  for (int i = 0; i < 8; ++i) x.u[i] = p[i];
  return x.v;
}

__device__ __forceinline__ float wave_max(float v) {
#pragma unroll
  for (int o = 16; o > 0; o >>= 1) v = fmaxf(v, __shfl_xor(v, o, 32));
  return v;
}
__device__ __forceinline__ float wave_sum(float v) {
#pragma unroll
  for (int o = 16; o > 0; o >>= 1) v += __shfl_xor(v, o, 32);
  return v;
}

__global__ void __launch_bounds__(256)
temporal_attention_fused(const float* __restrict__ node_feat,   // [B,172]
                         const float* __restrict__ time_feat,   // [B,100]
                         const float* __restrict__ edge_feat,   // [B,32,172]
                         const float* __restrict__ nbr_node,    // [B,32,172]
                         const float* __restrict__ nbr_time,    // [B,32,100]
                         const int*   __restrict__ nbr_mask,    // [B,32]
                         const float* __restrict__ Wq,          // [128,272]
                         const float* __restrict__ bq,          // [128]
                         const _Float16* __restrict__ WkvH,     // [256,448] f16
                         const float* __restrict__ bkv,         // [256]
                         const float* __restrict__ Wo,          // [128,128]
                         const float* __restrict__ bo,          // [128]
                         const float* __restrict__ gamma,
                         const float* __restrict__ beta,
                         float* __restrict__ out) {             // [B,128]
  extern __shared__ char smem[];
  float*    Z  = (float*)smem;                         // [128][ZS]
  _Float16* As = (_Float16*)(smem + Z_BYTES);          // [128][ASTR]
  _Float16* Bs = (_Float16*)(smem + Z_BYTES + AS_BYTES); // [256][BSTR]
  // post-GEMM aliases over the staging region:
  float* qin    = (float*)(smem + Z_BYTES);            // [4][272]
  float* qbuf   = (float*)(smem + Z_BYTES + 4352);     // [4][128]
  float* obuf   = (float*)(smem + Z_BYTES + 6400);     // [4][128]
  float* outbuf = (float*)(smem + Z_BYTES + 8448);     // [4][128]

  const int t    = threadIdx.x;
  const int lane = t & 31;
  const int wid  = t >> 5;          // 0..7
  const int wr   = wid >> 2;        // 0..1  (row block of 64)
  const int wc   = wid & 3;         // 0..3  (col block of 64)
  const int half = lane >> 4;
  const int l16  = lane & 15;
  const int b0   = blockIdx.x * BPERWG;

  // ---------------- Phase 1: KV projection via WMMA ----------------
  v8f acc[4][4] = {};
  const int kq = half * 16;

  for (int kc = 0; kc < KKV_PAD / 32; ++kc) {
    const int kb = kc * 32;
    // stage A: 128 rows x 32 k (f32 gather + concat -> f16)
#pragma unroll
    for (int i = 0; i < 16; ++i) {
      int idx = i * 256 + t;              // 4096 elements
      int r = idx >> 5, k = idx & 31;
      int b = b0 + (r >> 5), n = r & 31;
      int g = kb + k;
      float v;
      if (g < NODE_D)            v = nbr_node[(b * NNBR + n) * NODE_D + g];
      else if (g < NODE_D+EDGE_D) v = edge_feat[(b * NNBR + n) * EDGE_D + (g - NODE_D)];
      else if (g < KKV)          v = nbr_time[(b * NNBR + n) * TIME_D + (g - NODE_D - EDGE_D)];
      else                       v = 0.0f;
      As[r * ASTR + k] = (_Float16)v;
    }
    // stage B: Bs[c][k] = WkvH[c][kb+k], c = t (0..255), vectorized 2 halfs
    {
      unsigned* dst = (unsigned*)(Bs + t * BSTR);
      const unsigned* src = (const unsigned*)(WkvH + t * KKV_PAD + kb);
#pragma unroll
      for (int i = 0; i < 16; ++i) dst[i] = src[i];
    }
    __syncthreads();

    v16h bf[4];
#pragma unroll
    for (int ni = 0; ni < 4; ++ni)
      bf[ni] = load_b_frag(Bs, wc * 64 + ni * 16 + l16, kq);
#pragma unroll
    for (int mi = 0; mi < 4; ++mi) {
      v16h af = load_a_frag(As, wr * 64 + mi * 16 + l16, half);
#pragma unroll
      for (int ni = 0; ni < 4; ++ni)
        acc[mi][ni] = __builtin_amdgcn_wmma_f32_16x16x32_f16(
            false, af, false, bf[ni], (short)0, acc[mi][ni], false, false);
    }
    __syncthreads();
  }

  // write Z tile (+bias) to LDS. C/D layout: VGPR j -> M = j + 8*half, N = l16
#pragma unroll
  for (int mi = 0; mi < 4; ++mi)
#pragma unroll
    for (int ni = 0; ni < 4; ++ni) {
      int col = wc * 64 + ni * 16 + l16;
      float bias = bkv[col];
#pragma unroll
      for (int j = 0; j < 8; ++j) {
        int row = wr * 64 + mi * 16 + j + 8 * half;
        Z[row * ZS + col] = acc[mi][ni][j] + bias;
      }
    }
  __syncthreads();

  // ---------------- Phase 2: Q projection (tiny; VALU) ----------------
  for (int i = t; i < BPERWG * KQ; i += 256) {
    int bl = i / KQ, g = i - bl * KQ;
    int b = b0 + bl;
    qin[i] = (g < NODE_D) ? node_feat[b * NODE_D + g]
                          : time_feat[b * TIME_D + (g - NODE_D)];
  }
  __syncthreads();
#pragma unroll
  for (int rep = 0; rep < 2; ++rep) {
    int o = rep * 256 + t;                 // 512 outputs
    int bl = o >> 7, c = o & 127;
    float s = bq[c];
    const float* x = qin + bl * KQ;
    const float* w = Wq + c * KQ;
    for (int g = 0; g < KQ; ++g) s += x[g] * w[g];
    qbuf[bl * OUTD + c] = s;
  }
  __syncthreads();

  // ---------------- Phase 3: attention (wave = (batch, head)) ----------------
  {
    const int bl = wid >> 1;               // 0..3
    const int h  = wid & 1;                // 0..1
    const int b  = b0 + bl;
    const int nn = lane;                   // neighbor (mixed axis, faithful)
    // K[b,h,nn,d] = Z[bl*32 + h*16 + nn/2][(nn&1)*64 + d]
    const int zr = bl * NNBR + h * 16 + (nn >> 1);
    const int ck = (nn & 1) * HEADD;
    const float* qv = qbuf + bl * OUTD + h * HEADD;
    float s = 0.0f;
    const float* krow = Z + zr * ZS + ck;
#pragma unroll
    for (int d = 0; d < HEADD; ++d) s += qv[d] * krow[d];
    s *= 0.125f;                           // HEADD^-0.5
    if (nbr_mask[b * NNBR + nn] == 0) s = -1e10f;
    float m  = wave_max(s);
    float p  = __expf(s - m);
    float ps = wave_sum(p);
    p /= ps;
    // O[b,h,d] = sum_nn p_nn * V,  V = Z[..][128 + (nn&1)*64 + d]
    const int d0 = lane, d1 = lane + 32;
    float o0 = 0.0f, o1 = 0.0f;
    for (int j = 0; j < NNBR; ++j) {
      float pj = __shfl(p, j, 32);
      const float* vrow = Z + (bl * NNBR + h * 16 + (j >> 1)) * ZS + OUTD + (j & 1) * HEADD;
      o0 += pj * vrow[d0];
      o1 += pj * vrow[d1];
    }
    obuf[bl * OUTD + h * HEADD + d0] = o0;
    obuf[bl * OUTD + h * HEADD + d1] = o1;
  }
  __syncthreads();

  // ---------------- Phase 4: output projection ----------------
#pragma unroll
  for (int rep = 0; rep < 2; ++rep) {
    int o = rep * 256 + t;
    int bl = o >> 7, c = o & 127;
    float s = bo[c];
    const float* x = obuf + bl * OUTD;
    const float* w = Wo + c * OUTD;
#pragma unroll 4
    for (int k = 0; k < OUTD; ++k) s += x[k] * w[k];
    outbuf[bl * OUTD + c] = s;
  }
  __syncthreads();

  // ---------------- Phase 5: LayerNorm (waves 0..3, one row each) ----------
  if (wid < BPERWG) {
    const int bl = wid;
    float x[4], s = 0.0f;
#pragma unroll
    for (int i = 0; i < 4; ++i) { x[i] = outbuf[bl * OUTD + lane + 32 * i]; s += x[i]; }
    float mu = wave_sum(s) * (1.0f / OUTD);
    float v = 0.0f;
#pragma unroll
    for (int i = 0; i < 4; ++i) { float d = x[i] - mu; v += d * d; }
    float var  = wave_sum(v) * (1.0f / OUTD);
    float rstd = rsqrtf(var + LN_EPS);
#pragma unroll
    for (int i = 0; i < 4; ++i) {
      int c = lane + 32 * i;
      out[(b0 + bl) * OUTD + c] = (x[i] - mu) * rstd * gamma[c] + beta[c];
    }
  }
}

// ---------------------------------------------------------------------------
extern "C" void kernel_launch(void* const* d_in, const int* in_sizes, int n_in,
                              void* d_out, int out_size, void* d_ws, size_t ws_size,
                              hipStream_t stream) {
  const float* node_feat = (const float*)d_in[0];
  const float* time_feat = (const float*)d_in[1];
  const float* edge_feat = (const float*)d_in[2];
  const float* nbr_node  = (const float*)d_in[3];
  const float* nbr_time  = (const float*)d_in[4];
  const int*   nbr_mask  = (const int*)  d_in[5];
  const float* Wq        = (const float*)d_in[6];
  const float* bq        = (const float*)d_in[7];
  const float* Wkv       = (const float*)d_in[8];
  const float* bkv       = (const float*)d_in[9];
  const float* Wo        = (const float*)d_in[10];
  const float* bo        = (const float*)d_in[11];
  const float* gamma     = (const float*)d_in[12];
  const float* beta      = (const float*)d_in[13];
  float*       out       = (float*)d_out;

  _Float16* WkvH = (_Float16*)d_ws;       // 256*448 f16 = 229376 B

  // weight conversion: 256*448 elements
  prep_wkv_f16<<<(256 * KKV_PAD) / 256, 256, 0, stream>>>(Wkv, WkvH);

  // allow >64KB dynamic LDS (gfx1250 WGP: up to 320KB per workgroup)
  (void)hipFuncSetAttribute((const void*)temporal_attention_fused,
                            hipFuncAttributeMaxDynamicSharedMemorySize,
                            SMEM_BYTES);

  temporal_attention_fused<<<BATCH / BPERWG, 256, SMEM_BYTES, stream>>>(
      node_feat, time_feat, edge_feat, nbr_node, nbr_time, nbr_mask,
      Wq, bq, WkvH, bkv, Wo, bo, gamma, beta, out);
}
